// Conv3d_21148418966221
// MI455X (gfx1250) — matrix operand order repaired and checked by reading, not compiled, not run on hardware
//
#include <hip/hip_runtime.h>

typedef __attribute__((ext_vector_type(16))) _Float16 v16h;
typedef __attribute__((ext_vector_type(8)))  float    v8f;

// Input (8,32,48,48,48) f32, NCDHW
#define CIN     32
#define IN_SY   48
#define IN_SZ   2304          // 48*48
#define IN_SC   110592        // 48^3
#define IN_SB   3538944       // 32*48^3
// Output (8,64,46,46,46) f32
#define ODIM    46
#define OUT_SH  46
#define OUT_SD  2116          // 46*46
#define OUT_SC  97336         // 46^3
#define OUT_SB  6229504       // 64*46^3

#define TY      10            // tile y extent: 8 output rows + 2 halo
#define TROW    18            // tile x extent: 16 output cols + 2 halo
#define TPOS    (3 * TY * TROW)   // 540 tile positions
#define TSTR    40            // halves per position (32 ci + 8 pad) = 80 B

// ---------------------------------------------------------------------------
// Prep: reorder fp32 weights into per-tap f16 A-fragments for
// v_wmma_f32_16x16x32_f16 with K = ci (32, exact — no padding).
// aw[((mt*27 + tap)*32 + lane)*16 + i] = W[m = mt*16+(lane&15)][ci(i,lane)][tap]
//   ci(i,lane) = i + (lane>=16 ? 16 : 0)   (contiguous K order, matches B)
// ---------------------------------------------------------------------------
__global__ void prep_weights_f16(const float* __restrict__ w,
                                 _Float16* __restrict__ aw) {
    int idx  = blockIdx.x * 256 + threadIdx.x;      // 4*27*32*16 = 55296 total
    int i    = idx & 15;
    int lane = (idx >> 4) & 31;
    int rest = idx >> 9;                            // mt*27 + tap
    int tap  = rest % 27;
    int mt   = rest / 27;
    int ci   = i + ((lane >= 16) ? 16 : 0);
    int m    = mt * 16 + (lane & 15);
    aw[idx]  = (_Float16)w[m * 864 + ci * 27 + tap];
}

// ---------------------------------------------------------------------------
// Tap-loop implicit GEMM. WG = 128 threads = 4 wave32:
//   wave = (M half: 2 x 16 out-channels) x (4 oh rows).
// WG output tile: 64 channels x (16 ow x 8 oh) at one (b, od).
// Each B fragment read from LDS feeds TWO WMMAs (2 M-tiles per wave). Per tap
// per wave: 4 A global_b128 + 8 B ds_b128 + 8 x v_wmma_f32_16x16x32_f16.
// Epilogue uses immediate-offset stores from two base pointers + v8f bias.
// ---------------------------------------------------------------------------
__global__ __launch_bounds__(128)
void conv3d_wmma_tap(const float* __restrict__ in,
                     const _Float16* __restrict__ aw,
                     const float* __restrict__ bias,
                     float* __restrict__ out) {
    __shared__ __align__(64) _Float16 tile[TPOS * TSTR];   // 43,200 B

    const int t    = threadIdx.x;
    const int lane = t & 31;
    const int wave = t >> 5;            // 0..3
    const int mg   = wave & 1;          // M half: channels mg*32 .. mg*32+31
    const int ng   = wave >> 1;         // N slice: oh rows ng*4 .. ng*4+3

    const int ow0 = blockIdx.x * 16;                 // 0,16,32
    const int oh0 = blockIdx.y * 8;                  // 0..40
    const int bz  = blockIdx.z;
    const int b   = bz / ODIM;
    const int od  = bz % ODIM;

    const float* inb = in + (size_t)b * IN_SB;

    // ---- Stage input tile: 540 positions x 32 ci, coalesced along x. ----
    // 128 threads x 5 slots; slots 0..3 always in-range (t+384 <= 511 < 540).
    int offp[5];
    #pragma unroll
    for (int j = 0; j < 5; ++j) {
        int p = t + j * 128;
        int pc = (p < TPOS) ? p : 0;
        int z = pc / (TY * TROW), rem = pc - z * (TY * TROW);
        int y = rem / TROW, x = rem - y * TROW;
        int yi = oh0 + y; if (yi > 47) yi = 47;   // clamped reads only feed
        int xi = ow0 + x; if (xi > 47) xi = 47;   // out-of-range outputs
        offp[j] = (od + z) * IN_SZ + yi * IN_SY + xi;
    }
    const bool act4 = (t + 512 < TPOS);
    for (int ci = 0; ci < CIN; ++ci) {
        const float* inc = inb + ci * IN_SC;
        #pragma unroll
        for (int j = 0; j < 4; ++j)
            tile[(t + j * 128) * TSTR + ci] = (_Float16)inc[offp[j]];
        if (act4)
            tile[(t + 512) * TSTR + ci] = (_Float16)inc[offp[4]];
    }
    __syncthreads();

    // ---- Hot loop: 27 taps x (2 M-tiles x 4 rows), fully unrolled. ----
    const int nl = lane & 15;
    const int k0 = (lane >= 16) ? 16 : 0;   // B layout: lane<16 -> K 0..15
    const char* bbase = (const char*)tile
                      + (size_t)(nl * TSTR + k0) * 2
                      + (size_t)(ng * 4) * TROW * (TSTR * 2);
    // Two consecutive M-tiles (mt = mg*2, mg*2+1); slices are 27*512 halves.
    const _Float16* aptr0 = aw + ((size_t)(mg * 2) * 27 * 32 + lane) * 16;
    const _Float16* aptr1 = aptr0 + (size_t)27 * 512;

    v8f acc0[4] = {v8f{}, v8f{}, v8f{}, v8f{}};   // M-tile mg*2
    v8f acc1[4] = {v8f{}, v8f{}, v8f{}, v8f{}};   // M-tile mg*2+1

    #pragma unroll
    for (int kd = 0; kd < 3; ++kd) {
        #pragma unroll
        for (int kh = 0; kh < 3; ++kh) {
            #pragma unroll
            for (int kw = 0; kw < 3; ++kw) {
                const int tap = kd * 9 + kh * 3 + kw;
                const v16h a0 = *(const v16h*)(aptr0 + (size_t)tap * 512);
                const v16h a1 = *(const v16h*)(aptr1 + (size_t)tap * 512);
                #pragma unroll
                for (int r = 0; r < 4; ++r) {
                    const v16h bv = *(const v16h*)(bbase +
                        (kd * (TY * TROW) + (kh + r) * TROW + kw) * (TSTR * 2));
                    acc0[r] = __builtin_amdgcn_wmma_f32_16x16x32_f16(
                                  false, a0, false, bv, (short)0, acc0[r],
                                  false, false);
                    acc1[r] = __builtin_amdgcn_wmma_f32_16x16x32_f16(
                                  false, a1, false, bv, (short)0, acc1[r],
                                  false, false);
                }
            }
        }
    }

    // ---- Epilogue: D layout — lane<16: M=rr, lane>=16: M=rr+8; N=lane&15. ----
    // One per-lane base per M-half; every store offset is a compile-time imm
    // (max 7*OUT_SC*4 + 3*OUT_SH*4 = 2,725,960 B < 8 MB IOFFSET range).
    const int ow     = ow0 + nl;
    const int ohbase = oh0 + ng * 4;
    const int mb     = mg * 32 + ((lane >= 16) ? 8 : 0);   // multiple of 8
    const v8f bv0 = *(const v8f*)(bias + mb);              // channels mb..mb+7
    const v8f bv1 = *(const v8f*)(bias + mb + 16);         // channels +16
    float* ob0 = out + (size_t)b * OUT_SB + (size_t)mb * OUT_SC
               + od * OUT_SD + ohbase * OUT_SH + ow;
    float* ob1 = ob0 + (size_t)16 * OUT_SC;
    if (ow < ODIM) {
        #pragma unroll
        for (int r = 0; r < 4; ++r) {
            if (ohbase + r < ODIM) {
                #pragma unroll
                for (int rr = 0; rr < 8; ++rr) {
                    ob0[(size_t)rr * OUT_SC + r * OUT_SH] = acc0[r][rr] + bv0[rr];
                    ob1[(size_t)rr * OUT_SC + r * OUT_SH] = acc1[r][rr] + bv1[rr];
                }
            }
        }
    }
}

// ---------------------------------------------------------------------------
extern "C" void kernel_launch(void* const* d_in, const int* in_sizes, int n_in,
                              void* d_out, int out_size, void* d_ws, size_t ws_size,
                              hipStream_t stream) {
    const float* in   = (const float*)d_in[0];   // (8,32,48,48,48) f32
    const float* wgt  = (const float*)d_in[1];   // (64,32,3,3,3)   f32
    const float* bias = (const float*)d_in[2];   // (64,)           f32
    float* out = (float*)d_out;                  // (8,64,46,46,46) f32
    _Float16* aw = (_Float16*)d_ws;              // 108 KB pre-ordered weights

    prep_weights_f16<<<216, 256, 0, stream>>>(wgt, aw);   // 55296 threads

    dim3 grid(3 /*ow tiles*/, 6 /*oh tiles*/, 8 * ODIM /*b,od*/);
    conv3d_wmma_tap<<<grid, 128, 0, stream>>>(in, aw, bias, out);
}
